// Organic_GRN_57664230916772
// MI455X (gfx1250) — matrix operand order repaired and checked by reading, not compiled, not run on hardware
//
#include <hip/hip_runtime.h>
#include <hip/hip_bf16.h>
#include <stdint.h>

// ---------------------------------------------------------------------------
// Problem constants (match reference)
// ---------------------------------------------------------------------------
#define NN      100000
#define EE      400000
#define HID     256
#define NODE_D  256
#define BOND_D  14
#define GG      512
#define DEPTH   3

#define KP_I    288   // 270 padded to multiple of 32
#define KP_H    256
#define KP_O    512

typedef __attribute__((ext_vector_type(16))) __bf16 v16bf;
typedef __attribute__((ext_vector_type(8)))  float  v8f;

union FragU { uint4 u[2]; v16bf v; };

// round-to-nearest-even fp32 -> bf16 (bit pattern in ushort)
__device__ __forceinline__ unsigned short f2bf(float f) {
    unsigned int u = __float_as_uint(f);
    unsigned int r = u + 0x7FFFu + ((u >> 16) & 1u);
    return (unsigned short)(r >> 16);
}

// ---------------------------------------------------------------------------
// Utility: fill float buffer with zeros
// ---------------------------------------------------------------------------
__global__ void fill_zero_f32(float* __restrict__ p, size_t n) {
    size_t i = (size_t)blockIdx.x * blockDim.x + threadIdx.x;
    if (i < n) p[i] = 0.0f;
}

// ---------------------------------------------------------------------------
// Pack weight matrix W (K x 256, row-major fp32) into WMMA B-fragment layout,
// bf16, padded to Kp rows.  Layout: Bp[(kt*16 + ntile)*512 + lane*16 + j]
// where lane half h, element j covers K = kt*32 + 8h + j (j<8), kt*32 + 16 + 8h + (j-8) (j>=8)
// ---------------------------------------------------------------------------
__global__ void pack_weights(const float* __restrict__ W, unsigned short* __restrict__ Bp,
                             int K, int Kp) {
    size_t total = (size_t)(Kp >> 5) * 16 * 512;
    size_t i = (size_t)blockIdx.x * blockDim.x + threadIdx.x;
    if (i >= total) return;
    int kt    = (int)(i / (16 * 512));
    int rem   = (int)(i % (16 * 512));
    int ntile = rem / 512;
    int q     = rem % 512;
    int lane  = q / 16;
    int j     = q % 16;
    int khalf = lane >> 4;
    int n     = lane & 15;
    int k     = kt * 32 + (j < 8 ? khalf * 8 + j : 16 + khalf * 8 + (j - 8));
    int col   = ntile * 16 + n;
    float v   = (k < K) ? W[(size_t)k * 256 + col] : 0.0f;
    Bp[i] = f2bf(v);
}

// ---------------------------------------------------------------------------
// Build edge input matrix A (E x KP_I, bf16): concat(x[src[e]], edge_attr[e])
// ---------------------------------------------------------------------------
__global__ void build_A_edge(const float* __restrict__ x, const float* __restrict__ eattr,
                             const int* __restrict__ src, unsigned short* __restrict__ A) {
    size_t i = (size_t)blockIdx.x * blockDim.x + threadIdx.x;
    size_t total = (size_t)EE * KP_I;
    if (i >= total) return;
    int e = (int)(i / KP_I);
    int k = (int)(i % KP_I);
    float v;
    if (k < NODE_D)            v = x[(size_t)src[e] * NODE_D + k];
    else if (k < NODE_D + BOND_D) v = eattr[(size_t)e * BOND_D + (k - NODE_D)];
    else                       v = 0.0f;
    A[i] = f2bf(v);
}

// ---------------------------------------------------------------------------
// WMMA GEMM: out = relu(A(bf16, rows x Kp) @ Bpack + bias [+ addend]);
// optionally also writes pre-activation (outPre).
// One wave per block; each wave computes a 16-row x 64-col strip.
// gridDim = (rows/16, 4);  blockDim = 32.
// ---------------------------------------------------------------------------
__global__ __launch_bounds__(32)
void gemm_bf16_wmma(const unsigned short* __restrict__ A, int Kp,
                    const unsigned short* __restrict__ Bp,
                    const float* __restrict__ bias,
                    const float* __restrict__ addend,   // nullable
                    float* __restrict__ outRelu,
                    float* __restrict__ outPre) {       // nullable
    const int lane  = threadIdx.x;        // 0..31, full wave (EXEC all ones)
    const int khalf = lane >> 4;
    const int mn    = lane & 15;
    const int row0  = blockIdx.x * 16;
    const int ntg   = blockIdx.y;         // group of 4 n-tiles

    v8f acc0 = {}, acc1 = {}, acc2 = {}, acc3 = {};
    const int nKT = Kp >> 5;

    const unsigned short* arow = A + (size_t)(row0 + mn) * Kp + khalf * 8;

    for (int kt = 0; kt < nKT; ++kt) {
        FragU af;
        const unsigned short* ab = arow + kt * 32;
        af.u[0] = *(const uint4*)(ab);        // K = kt*32 + khalf*8 + 0..7
        af.u[1] = *(const uint4*)(ab + 16);   // K = kt*32 + 16 + khalf*8 + 0..7

        const unsigned short* bb =
            Bp + ((size_t)(kt * 16 + ntg * 4) * 512) + lane * 16;
        FragU b0, b1, b2, b3;
        b0.u[0] = *(const uint4*)(bb);            b0.u[1] = *(const uint4*)(bb + 8);
        b1.u[0] = *(const uint4*)(bb + 512);      b1.u[1] = *(const uint4*)(bb + 520);
        b2.u[0] = *(const uint4*)(bb + 1024);     b2.u[1] = *(const uint4*)(bb + 1032);
        b3.u[0] = *(const uint4*)(bb + 1536);     b3.u[1] = *(const uint4*)(bb + 1544);

        acc0 = __builtin_amdgcn_wmma_f32_16x16x32_bf16(false, af.v, false, b0.v, (short)0, acc0, false, false);
        acc1 = __builtin_amdgcn_wmma_f32_16x16x32_bf16(false, af.v, false, b1.v, (short)0, acc1, false, false);
        acc2 = __builtin_amdgcn_wmma_f32_16x16x32_bf16(false, af.v, false, b2.v, (short)0, acc2, false, false);
        acc3 = __builtin_amdgcn_wmma_f32_16x16x32_bf16(false, af.v, false, b3.v, (short)0, acc3, false, false);
    }

    // Epilogue.  C layout: lanes 0-15 -> N=lane, VGPR r -> M=r;
    //            lanes 16-31 -> N=lane-16, VGPR r -> M=8+r.
    v8f accs[4] = {acc0, acc1, acc2, acc3};
#pragma unroll
    for (int t = 0; t < 4; ++t) {
        int col = (ntg * 4 + t) * 16 + mn;
        float b = bias[col];
#pragma unroll
        for (int r = 0; r < 8; ++r) {
            int m = khalf * 8 + r;
            size_t idx = (size_t)(row0 + m) * HID + col;
            float v = accs[t][r] + b;
            if (addend) v += addend[idx];
            if (outPre) outPre[idx] = v;
            outRelu[idx] = v > 0.0f ? v : 0.0f;
        }
    }
}

// ---------------------------------------------------------------------------
// Segment-sum scatter: Mnode[dst[e]] += H[e]   (per element, atomic f32)
// ---------------------------------------------------------------------------
__global__ void scatter_add_edges(const float* __restrict__ H, const int* __restrict__ dst,
                                  float* __restrict__ Mnode) {
    size_t i = (size_t)blockIdx.x * blockDim.x + threadIdx.x;
    size_t total = (size_t)EE * HID;
    if (i >= total) return;
    int e = (int)(i >> 8);
    int k = (int)(i & 255);
    atomicAdd(&Mnode[(size_t)dst[e] * HID + k], H[i]);
}

// ---------------------------------------------------------------------------
// Edge message: Mbf[e] = bf16( Mnode[src[e]] - H[rev[e]] )
// ---------------------------------------------------------------------------
__global__ void edge_message(const float* __restrict__ Mnode, const float* __restrict__ H,
                             const int* __restrict__ src, const int* __restrict__ rev,
                             unsigned short* __restrict__ Mbf) {
    size_t i = (size_t)blockIdx.x * blockDim.x + threadIdx.x;
    size_t total = (size_t)EE * HID;
    if (i >= total) return;
    int e = (int)(i >> 8);
    int k = (int)(i & 255);
    float v = Mnode[(size_t)src[e] * HID + k] - H[(size_t)rev[e] * HID + k];
    Mbf[i] = f2bf(v);
}

// ---------------------------------------------------------------------------
// Build readout input (N x 512, bf16): concat(x, where(rowsum(M)==0, x, M)).
// One wave per node row; 8 rows per 256-thread block.
// ---------------------------------------------------------------------------
__global__ void build_A_node(const float* __restrict__ x, const float* __restrict__ Mnode,
                             unsigned short* __restrict__ An) {
    int wid  = threadIdx.x >> 5;
    int lane = threadIdx.x & 31;
    int row  = blockIdx.x * 8 + wid;
    if (row >= NN) return;
    float mv[8];
    float s = 0.0f;
#pragma unroll
    for (int j = 0; j < 8; ++j) {
        mv[j] = Mnode[(size_t)row * HID + lane + 32 * j];
        s += mv[j];
    }
#pragma unroll
    for (int off = 16; off > 0; off >>= 1) s += __shfl_xor(s, off, 32);
    bool useX = (s == 0.0f);
#pragma unroll
    for (int j = 0; j < 8; ++j) {
        int col = lane + 32 * j;
        float xv = x[(size_t)row * NODE_D + col];
        An[(size_t)row * KP_O + col]          = f2bf(xv);
        An[(size_t)row * KP_O + NODE_D + col] = f2bf(useX ? xv : mv[j]);
    }
}

// ---------------------------------------------------------------------------
// Pooling
// ---------------------------------------------------------------------------
__global__ void pool_count(const int* __restrict__ batch1, float* __restrict__ cnt) {
    int n = blockIdx.x * blockDim.x + threadIdx.x;
    if (n >= NN) return;
    atomicAdd(&cnt[batch1[n]], 1.0f);
}

__global__ void pool_sum(const float* __restrict__ Hn, const int* __restrict__ batch1,
                         float* __restrict__ acc) {
    size_t i = (size_t)blockIdx.x * blockDim.x + threadIdx.x;
    size_t total = (size_t)NN * HID;
    if (i >= total) return;
    int n = (int)(i >> 8);
    int k = (int)(i & 255);
    atomicAdd(&acc[(size_t)batch1[n] * HID + k], Hn[i]);
}

__global__ void pool_div(const float* __restrict__ acc, const float* __restrict__ cnt,
                         float* __restrict__ out) {
    int i = blockIdx.x * blockDim.x + threadIdx.x;
    if (i >= GG * HID) return;
    float c = cnt[i >> 8];
    out[i] = acc[i] / (c > 1.0f ? c : 1.0f);
}

// ---------------------------------------------------------------------------
// Host-side orchestration
// ---------------------------------------------------------------------------
extern "C" void kernel_launch(void* const* d_in, const int* in_sizes, int n_in,
                              void* d_out, int out_size, void* d_ws, size_t ws_size,
                              hipStream_t stream) {
    const float* x     = (const float*)d_in[0];
    const float* eattr = (const float*)d_in[1];
    const int*   eidx  = (const int*)d_in[2];     // (2, E) flat: row0=src, row1=dst
    const int*   rev   = (const int*)d_in[3];
    const int*   batch = (const int*)d_in[4];
    const float* Wi_w  = (const float*)d_in[5];
    const float* Wi_b  = (const float*)d_in[6];
    const float* Wh_w  = (const float*)d_in[7];
    const float* Wh_b  = (const float*)d_in[8];
    const float* Wo_w  = (const float*)d_in[9];
    const float* Wo_b  = (const float*)d_in[10];
    const int* src = eidx;
    const int* dst = eidx + EE;

    float* Hn_out   = (float*)d_out;              // N x 256
    float* pool_out = (float*)d_out + (size_t)NN * HID;

    // workspace layout (256B aligned chunks)
    char* base = (char*)d_ws;
    size_t off = 0;
    auto carve = [&](size_t bytes) -> char* {
        char* p = base + off;
        off += (bytes + 255) & ~(size_t)255;
        return p;
    };
    float* H0    = (float*)carve((size_t)EE * HID * 4);
    float* H     = (float*)carve((size_t)EE * HID * 4);
    float* Mnode = (float*)carve((size_t)NN * HID * 4);
    // shared region: A_edge (E x 288 bf16) / Mbf (E x 256 bf16) / A_node (N x 512 bf16)
    char*  shreg = carve((size_t)EE * KP_I * 2);
    unsigned short* Aedge = (unsigned short*)shreg;
    unsigned short* Mbf   = (unsigned short*)shreg;
    unsigned short* Anode = (unsigned short*)shreg;
    unsigned short* BpI = (unsigned short*)carve((size_t)(KP_I / 32) * 16 * 512 * 2);
    unsigned short* BpH = (unsigned short*)carve((size_t)(KP_H / 32) * 16 * 512 * 2);
    unsigned short* BpO = (unsigned short*)carve((size_t)(KP_O / 32) * 16 * 512 * 2);
    float* poolAcc = (float*)carve((size_t)GG * HID * 4);
    float* cnt     = (float*)carve((size_t)GG * 4);
    (void)ws_size; (void)in_sizes; (void)n_in; (void)out_size;

    const int T = 256;
    auto blocks = [](size_t n, int t) { return (unsigned)((n + t - 1) / t); };

    // 1) pack weights into WMMA fragment layout (bf16)
    pack_weights<<<blocks((size_t)(KP_I/32)*16*512, T), T, 0, stream>>>(Wi_w, BpI, NODE_D + BOND_D, KP_I);
    pack_weights<<<blocks((size_t)(KP_H/32)*16*512, T), T, 0, stream>>>(Wh_w, BpH, KP_H, KP_H);
    pack_weights<<<blocks((size_t)(KP_O/32)*16*512, T), T, 0, stream>>>(Wo_w, BpO, KP_O, KP_O);

    // 2) build edge inputs, H0 = A @ Wi + b ; H = relu(H0)
    build_A_edge<<<blocks((size_t)EE * KP_I, T), T, 0, stream>>>(x, eattr, src, Aedge);
    gemm_bf16_wmma<<<dim3(EE / 16, 4), 32, 0, stream>>>(Aedge, KP_I, BpI, Wi_b,
                                                        nullptr, H, H0);

    // 3) message-passing iterations
    for (int it = 0; it < DEPTH - 1; ++it) {
        fill_zero_f32<<<blocks((size_t)NN * HID, T), T, 0, stream>>>(Mnode, (size_t)NN * HID);
        scatter_add_edges<<<blocks((size_t)EE * HID, T), T, 0, stream>>>(H, dst, Mnode);
        edge_message<<<blocks((size_t)EE * HID, T), T, 0, stream>>>(Mnode, H, src, rev, Mbf);
        gemm_bf16_wmma<<<dim3(EE / 16, 4), 32, 0, stream>>>(Mbf, KP_H, BpH, Wh_b,
                                                            H0, H, nullptr);
    }

    // 4) final node aggregation + readout
    fill_zero_f32<<<blocks((size_t)NN * HID, T), T, 0, stream>>>(Mnode, (size_t)NN * HID);
    scatter_add_edges<<<blocks((size_t)EE * HID, T), T, 0, stream>>>(H, dst, Mnode);
    build_A_node<<<blocks((size_t)NN, 8) , 256, 0, stream>>>(x, Mnode, Anode);
    gemm_bf16_wmma<<<dim3(NN / 16, 4), 32, 0, stream>>>(Anode, KP_O, BpO, Wo_b,
                                                        nullptr, Hn_out, nullptr);

    // 5) mean pooling over graphs
    fill_zero_f32<<<blocks((size_t)GG * HID, T), T, 0, stream>>>(poolAcc, (size_t)GG * HID);
    fill_zero_f32<<<blocks((size_t)GG, T), T, 0, stream>>>(cnt, (size_t)GG);
    pool_count<<<blocks((size_t)NN, T), T, 0, stream>>>(batch, cnt);
    pool_sum<<<blocks((size_t)NN * HID, T), T, 0, stream>>>(Hn_out, batch, poolAcc);
    pool_div<<<blocks((size_t)GG * HID, T), T, 0, stream>>>(poolAcc, cnt, pool_out);
}
